// MLP_GAT_Block_67628555043538
// MI455X (gfx1250) — matrix operand order repaired and checked by reading, not compiled, not run on hardware
//
#include <hip/hip_runtime.h>

typedef __bf16 bf16;
typedef __bf16 v16bf __attribute__((ext_vector_type(16)));
typedef float  v8f   __attribute__((ext_vector_type(8)));

constexpr float kAlpha  = 0.2f;
constexpr float kNegInf = -9e15f;

__device__ __forceinline__ float elu1(float x) { return x > 0.f ? x : (__expf(x) - 1.f); }

union FragBF { v16bf v; bf16 e[16]; uint4 q[2]; };
union Pack8  { uint4 q; bf16 e[8]; };

// ---------------------------------------------------------------------------
// fp32 -> bf16 elementwise (x)
// ---------------------------------------------------------------------------
__global__ void k_f32_to_bf16(const float* __restrict__ s, bf16* __restrict__ d, int n) {
  int i = blockIdx.x * blockDim.x + threadIdx.x;
  int stride = gridDim.x * blockDim.x;
  for (; i < n; i += stride) d[i] = (bf16)s[i];
}

// ---------------------------------------------------------------------------
// fp32 [K,Nf] -> bf16 transposed [Nf,K]  (weights; tiny, done once)
// ---------------------------------------------------------------------------
__global__ void k_transpose_bf16(const float* __restrict__ s, bf16* __restrict__ d,
                                 int K, int Nf) {
  int idx = blockIdx.x * blockDim.x + threadIdx.x;
  int total = K * Nf;
  int stride = gridDim.x * blockDim.x;
  for (; idx < total; idx += stride) {
    int n = idx / K;
    int k = idx - n * K;
    d[idx] = (bf16)s[(size_t)k * Nf + n];
  }
}

// ---------------------------------------------------------------------------
// Dense GEMM: D = act(A[M,K]_bf16 @ B + bias), B pre-transposed BT[Nf,K].
// 256 threads = 8 waves; each wave register-blocks a 32x32 tile (2x2 WMMAs),
// block tile 128x64. Per k-step: 8 contiguous b128 loads -> 4 WMMAs, every
// fragment reused twice, 4 independent accumulator chains for ILP.
// Outputs: row-major f32/bf16 and/or batch-transposed bf16 outT[b][n][i]
// (packed 8-wide b128 stores, Nb = nodes per batch). nullptr skips a sink.
// ---------------------------------------------------------------------------
template <int ACT>
__global__ void k_gemm(const bf16* __restrict__ A, const bf16* __restrict__ BT,
                       const float* __restrict__ bias,
                       float* __restrict__ outF, bf16* __restrict__ outB,
                       bf16* __restrict__ outT,
                       int M, int K, int Nf, int Nb) {
  const int tid  = threadIdx.x;
  const int lane = tid & 31;
  const int w    = tid >> 5;
  const int m0   = blockIdx.y * 128 + (w >> 1) * 32;
  const int n0   = blockIdx.x * 64 + (w & 1) * 32;
  const int half = lane >> 4;
  const int l15  = lane & 15;

  const bf16* Ar[2] = { A + (size_t)(m0 + l15) * K, A + (size_t)(m0 + 16 + l15) * K };
  const bf16* Br[2] = { BT + (size_t)(n0 + l15) * K, BT + (size_t)(n0 + 16 + l15) * K };

  v8f acc[2][2] = {};
  for (int kb = 0; kb < K; kb += 32) {
    FragBF a[2], b[2];
#pragma unroll
    for (int t = 0; t < 2; ++t) {
      a[t].q[0] = *(const uint4*)(Ar[t] + kb + half * 8);
      a[t].q[1] = *(const uint4*)(Ar[t] + kb + 16 + half * 8);
      b[t].q[0] = *(const uint4*)(Br[t] + kb + half * 8);
      b[t].q[1] = *(const uint4*)(Br[t] + kb + 16 + half * 8);
    }
#pragma unroll
    for (int mi = 0; mi < 2; ++mi)
#pragma unroll
      for (int ni = 0; ni < 2; ++ni)
        acc[mi][ni] = __builtin_amdgcn_wmma_f32_16x16x32_bf16(
            false, a[mi].v, false, b[ni].v, (short)0, acc[mi][ni], false, false);
  }

#pragma unroll
  for (int mi = 0; mi < 2; ++mi) {
#pragma unroll
    for (int ni = 0; ni < 2; ++ni) {
      const int n  = n0 + ni * 16 + l15;
      const float bv = bias ? bias[n] : 0.f;
      float vals[8];
#pragma unroll
      for (int r = 0; r < 8; ++r) {
        float v = acc[mi][ni][r] + bv;
        if (ACT == 1) v = v > 0.f ? v : 0.f;
        vals[r] = v;
      }
      if (outF || outB) {
#pragma unroll
        for (int r = 0; r < 8; ++r) {
          size_t idx = (size_t)(m0 + mi * 16 + half * 8 + r) * Nf + n;
          if (outF) outF[idx] = vals[r];
          if (outB) outB[idx] = (bf16)vals[r];
        }
      }
      if (outT) {  // batch-transposed bf16: outT[(b*Nf + n)*Nb + i], 8 packed
        int mb = m0 + mi * 16 + half * 8;
        int bi = mb / Nb;
        int i  = mb - bi * Nb;
        Pack8 p;
#pragma unroll
        for (int r = 0; r < 8; ++r) p.e[r] = (bf16)vals[r];
        *(uint4*)(outT + ((size_t)bi * Nf + n) * Nb + i) = p.q;
      }
    }
  }
}

// ---------------------------------------------------------------------------
// From transposed h: f1[b,i] = sum_f hT[b][f][i]*a[f], f2 with a[F+f].
// Thread per node i (coalesced along N), loop over features.
// ---------------------------------------------------------------------------
__global__ void k_gat_vec(const bf16* __restrict__ hT, const float* __restrict__ a,
                          float* __restrict__ f1, float* __restrict__ f2,
                          int F, int N) {
  const int b = blockIdx.y;
  const int i = blockIdx.x * blockDim.x + threadIdx.x;
  const bf16* p = hT + (size_t)b * F * N + i;
  float s1 = 0.f, s2 = 0.f;
  for (int f = 0; f < F; ++f) {
    float x = (float)p[(size_t)f * N];
    s1 += x * a[f];
    s2 += x * a[F + f];
  }
  f1[(size_t)b * N + i] = s1;
  f2[(size_t)b * N + i] = s2;
}

// ---------------------------------------------------------------------------
// Per-row softmax stats: m = max_j e_ij, s = sum_j exp(e_ij - m)
// e_ij = lrelu(f1_i + f2_j), masked by adj (nullptr = unmasked).
// One wave per (b,i) row; N power of two.
// ---------------------------------------------------------------------------
__global__ void k_gat_stats(const float* __restrict__ f1, const float* __restrict__ f2,
                            const int* __restrict__ adj,
                            float* __restrict__ rmax, float* __restrict__ rsum, int N) {
  int row  = blockIdx.x * (blockDim.x >> 5) + (threadIdx.x >> 5);
  int lane = threadIdx.x & 31;
  int i    = row & (N - 1);
  int bb   = row - i;  // b*N
  float f1v = f1[row];
  const float* f2b = f2 + bb;
  const int* adjr  = adj ? adj + (size_t)i * N : nullptr;

  float m = kNegInf;
  for (int j = lane; j < N; j += 32) {
    float e = f1v + f2b[j];
    e = e > 0.f ? e : kAlpha * e;
    if (adjr && adjr[j] == 0) e = kNegInf;
    m = fmaxf(m, e);
  }
  for (int o = 16; o; o >>= 1) m = fmaxf(m, __shfl_xor(m, o, 32));
  if (m < -8e14f) m = 0.f;

  float s = 0.f;
  for (int j = lane; j < N; j += 32) {
    float e = f1v + f2b[j];
    e = e > 0.f ? e : kAlpha * e;
    if (adjr && adjr[j] == 0) e = kNegInf;
    s += __expf(e - m);
  }
  for (int o = 16; o; o >>= 1) s += __shfl_xor(s, o, 32);
  if (!lane) { rmax[row] = m; rsum[row] = fmaxf(s, 1e-30f); }
}

// ---------------------------------------------------------------------------
// Fused attention GEMM: out = epi( (P @ h) / rowsum + residual )
// A fragment (softmax numerators) built once per k-step in registers and
// reused across TWO 16-wide feature tiles (2 WMMAs) -> halves exp recompute.
// B fragments: contiguous b128 loads from transposed hT[b][f][j].
// EPI 0: elu(elu(v)) + res -> bf16 (gat1).  EPI 1: relu(elu(elu(v)) + res) -> f32 (gat2).
// ---------------------------------------------------------------------------
template <int EPI>
__global__ void k_gat_attn(const bf16* __restrict__ hT, const float* __restrict__ f1,
                           const float* __restrict__ f2, const int* __restrict__ adj,
                           const float* __restrict__ rmax, const float* __restrict__ rsum,
                           const float* __restrict__ res,
                           float* __restrict__ outF, bf16* __restrict__ outB,
                           int N, int F) {
  const int b    = blockIdx.z;
  const int tid  = threadIdx.x;
  const int lane = tid & 31;
  const int w    = tid >> 5;
  const int i0   = blockIdx.y * 64 + (w >> 1) * 16;
  const int f0   = blockIdx.x * 64 + (w & 1) * 32;  // wave covers f0..f0+31
  const int half = lane >> 4;
  const int l15  = lane & 15;

  const int irow   = i0 + l15;                  // A-matrix row for this lane
  const size_t rg0 = (size_t)b * N + irow;
  const float f1v  = f1[rg0];
  const float mrow = rmax[rg0];
  const float* f2b = f2 + (size_t)b * N;
  const int* adjr  = adj ? adj + (size_t)irow * N : nullptr;
  const bf16* hr[2] = { hT + ((size_t)b * F + f0 + l15) * N,
                        hT + ((size_t)b * F + f0 + 16 + l15) * N };

  v8f acc[2] = {};
  for (int kb = 0; kb < N; kb += 32) {
    // --- A fragment: softmax numerators, bf16 (built once, used twice) ---
    const int j0 = kb + half * 8;        // e[0..7]  -> K = j0..j0+7
    const int j1 = kb + 16 + half * 8;   // e[8..15] -> K = j1..j1+7
    float fv[16];
    *(float4*)&fv[0]  = *(const float4*)(f2b + j0);
    *(float4*)&fv[4]  = *(const float4*)(f2b + j0 + 4);
    *(float4*)&fv[8]  = *(const float4*)(f2b + j1);
    *(float4*)&fv[12] = *(const float4*)(f2b + j1 + 4);
    int mv[16];
    if (adjr) {
      *(int4*)&mv[0]  = *(const int4*)(adjr + j0);
      *(int4*)&mv[4]  = *(const int4*)(adjr + j0 + 4);
      *(int4*)&mv[8]  = *(const int4*)(adjr + j1);
      *(int4*)&mv[12] = *(const int4*)(adjr + j1 + 4);
    } else {
#pragma unroll
      for (int t = 0; t < 16; ++t) mv[t] = 1;
    }
    FragBF a;
#pragma unroll
    for (int t = 0; t < 16; ++t) {
      float e = f1v + fv[t];
      e = e > 0.f ? e : kAlpha * e;
      float p = (mv[t] == 0) ? 0.f : __expf(e - mrow);
      a.e[t] = (bf16)p;
    }

    FragBF bfr[2];
#pragma unroll
    for (int t = 0; t < 2; ++t) {
      bfr[t].q[0] = *(const uint4*)(hr[t] + kb + half * 8);
      bfr[t].q[1] = *(const uint4*)(hr[t] + kb + 16 + half * 8);
    }
#pragma unroll
    for (int t = 0; t < 2; ++t)
      acc[t] = __builtin_amdgcn_wmma_f32_16x16x32_bf16(false, a.v, false, bfr[t].v,
                                                       (short)0, acc[t], false, false);
  }

#pragma unroll
  for (int r = 0; r < 8; ++r) {
    int m = i0 + half * 8 + r;
    size_t rg = (size_t)b * N + m;
    float inv = 1.f / rsum[rg];
#pragma unroll
    for (int fi = 0; fi < 2; ++fi) {
      float v = acc[fi][r] * inv;
      v = elu1(elu1(v));
      size_t idx = rg * F + f0 + fi * 16 + l15;
      v += res[idx];
      if (EPI == 1) v = v > 0.f ? v : 0.f;
      if (outF) outF[idx] = v;
      if (outB) outB[idx] = (bf16)v;
    }
  }
}

// ---------------------------------------------------------------------------
extern "C" void kernel_launch(void* const* d_in, const int* in_sizes, int n_in,
                              void* d_out, int out_size, void* d_ws, size_t ws_size,
                              hipStream_t stream) {
  (void)in_sizes; (void)n_in; (void)out_size; (void)ws_size;
  constexpr int B = 16, N = 1024, Fin = 256, H = 512, O = 256;
  constexpr int M = B * N;

  const float* x      = (const float*)d_in[0];
  const int*   adj    = (const int*)d_in[1];
  const float* w1     = (const float*)d_in[2];
  const float* b1     = (const float*)d_in[3];
  const float* w2     = (const float*)d_in[4];
  const float* b2     = (const float*)d_in[5];
  const float* wfc    = (const float*)d_in[6];
  const float* a_fc   = (const float*)d_in[7];
  const float* wh     = (const float*)d_in[8];
  const float* a_hand = (const float*)d_in[9];
  float* out = (float*)d_out;

  char* ws = (char*)d_ws;
  size_t off = 0;
  auto alloc = [&](size_t bytes) -> char* {
    char* p = ws + off;
    off += (bytes + 255) & ~(size_t)255;
    return p;
  };
  bf16* xbf   = (bf16*)alloc((size_t)M * Fin * 2);  // row-major A for mlp1
  bf16* w1T   = (bf16*)alloc((size_t)Fin * H * 2);  // [H,Fin]
  bf16* wfcT  = (bf16*)alloc((size_t)H * H * 2);    // [H,H]
  bf16* w2T   = (bf16*)alloc((size_t)H * O * 2);    // [O,H]
  bf16* whT   = (bf16*)alloc((size_t)O * O * 2);    // [O,O]
  float* t0   = (float*)alloc((size_t)M * H * 4);   // mlp1 out / residual1; reused as t1
  bf16* t0bf  = (bf16*)alloc((size_t)M * H * 2);    // row-major; reused as t1bf
  bf16* h1T   = (bf16*)alloc((size_t)M * H * 2);    // transposed h [b][f][i]; reused as h2T
  bf16* o1bf  = (bf16*)alloc((size_t)M * H * 2);    // gat1 block out (mlp2 A operand)
  float* f1   = (float*)alloc((size_t)M * 4);
  float* f2   = (float*)alloc((size_t)M * 4);
  float* rm   = (float*)alloc((size_t)M * 4);
  float* rs   = (float*)alloc((size_t)M * 4);
  float* t1   = t0;    // alias: t0 dead after gat1 attn
  bf16*  t1bf = t0bf;  // alias
  bf16*  h2T  = h1T;   // alias: h1 dead after gat1 attn

  {  // conversions / weight transposes
    int n = M * Fin, g = (n + 255) / 256; if (g > 4096) g = 4096;
    k_f32_to_bf16<<<g, 256, 0, stream>>>(x, xbf, n);
    k_transpose_bf16<<<(Fin * H + 255) / 256, 256, 0, stream>>>(w1, w1T, Fin, H);
    k_transpose_bf16<<<(H * H + 255) / 256, 256, 0, stream>>>(wfc, wfcT, H, H);
    k_transpose_bf16<<<(H * O + 255) / 256, 256, 0, stream>>>(w2, w2T, H, O);
    k_transpose_bf16<<<(O * O + 255) / 256, 256, 0, stream>>>(wh, whT, O, O);
  }

  // mlp1: t0 = relu(x @ w1 + b1)  (relu idempotent -> t0 also feeds gat1)
  k_gemm<1><<<dim3(H / 64, M / 128), 256, 0, stream>>>(
      xbf, w1T, b1, t0, t0bf, nullptr, M, Fin, H, N);
  // gat1 h (stored transposed): h1T = (t0 @ W_fc)^T per batch
  k_gemm<0><<<dim3(H / 64, M / 128), 256, 0, stream>>>(
      t0bf, wfcT, nullptr, nullptr, nullptr, h1T, M, H, H, N);
  // gat1 attention scalars + stats (unmasked)
  k_gat_vec<<<dim3(N / 256, B), 256, 0, stream>>>(h1T, a_fc, f1, f2, H, N);
  k_gat_stats<<<M / 8, 256, 0, stream>>>(f1, f2, nullptr, rm, rs, N);
  // gat1 fused P@h + elu(elu())+residual -> o1bf (row-major)
  k_gat_attn<0><<<dim3(H / 64, N / 64, B), 256, 0, stream>>>(
      h1T, f1, f2, nullptr, rm, rs, t0, nullptr, o1bf, N, H);

  // mlp2: t1 = relu(o1 @ w2 + b2)
  k_gemm<1><<<dim3(O / 64, M / 128), 256, 0, stream>>>(
      o1bf, w2T, b2, t1, t1bf, nullptr, M, H, O, N);
  // gat2 h (transposed): h2T = (t1 @ W_hand)^T per batch
  k_gemm<0><<<dim3(O / 64, M / 128), 256, 0, stream>>>(
      t1bf, whT, nullptr, nullptr, nullptr, h2T, M, O, O, N);
  // gat2 attention scalars + masked stats
  k_gat_vec<<<dim3(N / 256, B), 256, 0, stream>>>(h2T, a_hand, f1, f2, O, N);
  k_gat_stats<<<M / 8, 256, 0, stream>>>(f1, f2, adj, rm, rs, N);
  // gat2 fused attention + relu(elu(elu())+residual) -> d_out (f32)
  k_gat_attn<1><<<dim3(O / 64, N / 64, B), 256, 0, stream>>>(
      h2T, f1, f2, adj, rm, rs, t1, out, nullptr, N, O);
}
